// HACDLayer_86792699117877
// MI455X (gfx1250) — matrix operand order repaired and checked by reading, not compiled, not run on hardware
//
#include <hip/hip_runtime.h>
#include <stdint.h>

#define N_NODES 50000
#define N_EDGES 400000
#define DIN     256
#define HEADS   4
#define FDIM    64
#define HF      256
#define PMETA   3
#define HID     128
#define NEG     0.2f
#define MTOT    (PMETA * N_NODES)   // 150000

typedef __attribute__((ext_vector_type(16))) _Float16 v16h;
typedef __attribute__((ext_vector_type(2)))  __fp16   v2fp;   // cvt_pkrtz return type
typedef __attribute__((ext_vector_type(8)))  float    v8f;

// ---------------- workspace layout (bytes) ----------------
#define XW_OFF    ((size_t)0)
#define XW_BYTES  ((size_t)PMETA * N_NODES * HF * 4)          // 153,600,000
#define Z_OFF     (XW_OFF + XW_BYTES)
#define AS_OFF    (Z_OFF + XW_BYTES)
#define PNH_BYTES ((size_t)PMETA * N_NODES * HEADS * 4)       // 2,400,000
#define AD_OFF    (AS_OFF + PNH_BYTES)
#define M_OFF     (AD_OFF + PNH_BYTES)
#define S_OFF     (M_OFF + PNH_BYTES)
#define WPK_OFF   (S_OFF + PNH_BYTES)
#define WPK_BYTES ((size_t)PMETA * HF * 128 * 4)              // 393,216
#define W1PK_OFF  (WPK_OFF + WPK_BYTES)
#define W1PK_BYTES ((size_t)HID * 128 * 4)                    // 65,536
#define WSUM_OFF  (W1PK_OFF + W1PK_BYTES)
#define BETA_OFF  (WSUM_OFF + 256)

// ---------------- WMMA fragment helpers ----------------
// A fragment (16x32 f16) from a pre-clamped f32 row pointer.
// lane<16 -> K chunks [0..7],[16..23]; lane>=16 -> [8..15],[24..31] (klo folded by caller).
__device__ inline v16h load_a_f32(const float* __restrict__ arow, int k0) {
  float tmp[16];
  *(float4*)(tmp + 0)  = *(const float4*)(arow + k0 + 0);
  *(float4*)(tmp + 4)  = *(const float4*)(arow + k0 + 4);
  *(float4*)(tmp + 8)  = *(const float4*)(arow + k0 + 16);
  *(float4*)(tmp + 12) = *(const float4*)(arow + k0 + 20);
  union { v16h v; v2fp h2[8]; } a;
#pragma unroll
  for (int j = 0; j < 8; j++)
    a.h2[j] = __builtin_amdgcn_cvt_pkrtz(tmp[2 * j], tmp[2 * j + 1]);
  return a.v;
}

// B fragment (32x16 f16) from pre-packed weights: wb[col][k2] holds f16 pair (K=2*k2, 2*k2+1).
// lane<16 -> K 0..7,16..23 of the k-step, lane>=16 -> K 8..15,24..31; col = (lane&15) within tile.
__device__ inline v16h load_b_pk(const uint32_t* __restrict__ wb, int col, int kb2) {
  union { v16h v; uint4 q[2]; } b;
  const uint32_t* p = wb + (size_t)col * 128 + kb2;
  b.q[0] = *(const uint4*)(p);
  b.q[1] = *(const uint4*)(p + 8);
  return b.v;
}

// One k-step: 8 WMMA tiles with a distance-2 software pipeline on B fragments,
// so each WMMA overlaps the loads of the tile two slots ahead.
__device__ inline void gemm_step(const v16h a, const uint32_t* __restrict__ wb,
                                 int colbase, int kb2, v8f* acc) {
  v16h bq0 = load_b_pk(wb, colbase + 0 * 16, kb2);
  v16h bq1 = load_b_pk(wb, colbase + 1 * 16, kb2);
#pragma unroll
  for (int nt = 0; nt < 8; nt++) {
    v16h bcur = (nt & 1) ? bq1 : bq0;
    if (nt + 2 < 8) {
      v16h bn = load_b_pk(wb, colbase + (nt + 2) * 16, kb2);
      if (nt & 1) bq1 = bn; else bq0 = bn;
    }
    acc[nt] = __builtin_amdgcn_wmma_f32_16x16x32_f16(
        false, a, false, bcur, (short)0, acc[nt], false, false);
  }
}

// ---------------- kernels ----------------

// Pack gat_W [P][DIN][HF] and sa_w1 [DIN][HID] into f16-pair B layout; zero wsum.
__global__ void k_pack(const float* __restrict__ gatW, const float* __restrict__ saw1,
                       uint32_t* __restrict__ wpk, uint32_t* __restrict__ w1pk,
                       float* __restrict__ wsum) {
  int gid = blockIdx.x * blockDim.x + threadIdx.x;
  if (gid < 3) wsum[gid] = 0.f;
  if (gid < PMETA * HF * 128) {
    int p = gid / (HF * 128);
    int r = gid % (HF * 128);
    int n = r / 128, k2 = r % 128;
    union { v2fp h; uint32_t u; } pk;
    pk.h = __builtin_amdgcn_cvt_pkrtz(gatW[((size_t)p * DIN + 2 * k2) * HF + n],
                                      gatW[((size_t)p * DIN + 2 * k2 + 1) * HF + n]);
    wpk[gid] = pk.u;
  } else {
    int g = gid - PMETA * HF * 128;
    if (g < HID * 128) {
      int n = g / 128, k2 = g % 128;
      union { v2fp h; uint32_t u; } pk;
      pk.h = __builtin_amdgcn_cvt_pkrtz(saw1[(size_t)(2 * k2) * HID + n],
                                        saw1[(size_t)(2 * k2 + 1) * HID + n]);
      w1pk[(size_t)n * 128 + k2] = pk.u;
    }
  }
}

// xw[p] = h @ W[p] via WMMA. grid = (ceil(N/128), 6); blockIdx.y = p*2 + coltile.
__global__ void __launch_bounds__(256)
k_gemm_xw(const float* __restrict__ hmat, const uint32_t* __restrict__ wpk,
          float* __restrict__ xw) {
  const int wid = threadIdx.x >> 5;
  const int lane = threadIdx.x & 31;
  const int p = blockIdx.y >> 1;
  const int ncol0 = (blockIdx.y & 1) * 128;
  const int m0 = blockIdx.x * 128 + wid * 16;
  const uint32_t* wb = wpk + (size_t)p * (HF * 128);

  int arow_i = m0 + (lane & 15);
  arow_i = arow_i < N_NODES ? arow_i : N_NODES - 1;   // clamp: OOB rows never stored
  const float* arow = hmat + (size_t)arow_i * DIN + ((lane < 16) ? 0 : 8);
  const int klo2 = (lane < 16) ? 0 : 4;               // B k2 offset for this lane half

  v8f acc[8] = {};
  // k-loop unrolled x2 with independent A fragments (avoids WMMA->VALU WAR nops)
  for (int k0 = 0; k0 < DIN; k0 += 64) {
    v16h a0 = load_a_f32(arow, k0);
    gemm_step(a0, wb, ncol0, (k0 >> 1) + klo2, acc);
    v16h a1 = load_a_f32(arow, k0 + 32);
    gemm_step(a1, wb, ncol0, ((k0 + 32) >> 1) + klo2, acc);
  }

  float* out = xw + (size_t)p * N_NODES * HF;
  const int rbase = m0 + ((lane < 16) ? 0 : 8);
#pragma unroll
  for (int nt = 0; nt < 8; nt++) {
    int col = ncol0 + nt * 16 + (lane & 15);
#pragma unroll
    for (int i = 0; i < 8; i++) {
      int row = rbase + i;
      if (row < N_NODES) out[(size_t)row * HF + col] = acc[nt][i];
    }
  }
}

// alpha_s/alpha_d[p][n][h] = dot(xw[p][n][h*64: ], att_src/dst[p][h])
__global__ void k_alpha(const float* __restrict__ xw, const float* __restrict__ asrc,
                        const float* __restrict__ adst, float* __restrict__ als,
                        float* __restrict__ ald) {
  int gid = blockIdx.x * blockDim.x + threadIdx.x;
  if (gid >= PMETA * N_NODES * HEADS) return;
  int p = gid / (N_NODES * HEADS);
  int r = gid % (N_NODES * HEADS);
  int n = r >> 2, hh = r & 3;
  const float* x = xw + ((size_t)p * N_NODES + n) * HF + hh * FDIM;
  const float* va = asrc + (size_t)(p * HEADS + hh) * FDIM;
  const float* vd = adst + (size_t)(p * HEADS + hh) * FDIM;
  float s0 = 0.f, s1 = 0.f;
#pragma unroll 4
  for (int f = 0; f < FDIM; f++) { float xv = x[f]; s0 += xv * va[f]; s1 += xv * vd[f]; }
  als[gid] = s0;
  ald[gid] = s1;
}

// m init with self-loop logit
__global__ void k_minit(const float* __restrict__ als, const float* __restrict__ ald,
                        float* __restrict__ m) {
  int gid = blockIdx.x * blockDim.x + threadIdx.x;
  if (gid >= PMETA * N_NODES * HEADS) return;
  float e = als[gid] + ald[gid];
  m[gid] = e > 0.f ? e : NEG * e;
}

// edge pass 1: atomic max of edge logits into m (per meta-path launch; pointers pre-offset)
__global__ void k_edge_max(const int* __restrict__ ei, const float* __restrict__ als,
                           const float* __restrict__ ald, float* __restrict__ m) {
  int e = blockIdx.x * blockDim.x + threadIdx.x;
  if (e >= N_EDGES) return;
  int src = ei[e], dst = ei[N_EDGES + e];
#pragma unroll
  for (int hh = 0; hh < HEADS; hh++) {
    float v = als[src * 4 + hh] + ald[dst * 4 + hh];
    v = v > 0.f ? v : NEG * v;
    float* addr = &m[dst * 4 + hh];
    if (v >= 0.f) atomicMax((int*)addr, __float_as_int(v));
    else          atomicMin((unsigned int*)addr, __float_as_uint(v));
  }
}

// s init with self-loop term
__global__ void k_sinit(const float* __restrict__ als, const float* __restrict__ ald,
                        const float* __restrict__ m, float* __restrict__ s) {
  int gid = blockIdx.x * blockDim.x + threadIdx.x;
  if (gid >= PMETA * N_NODES * HEADS) return;
  float e = als[gid] + ald[gid];
  e = e > 0.f ? e : NEG * e;
  s[gid] = __expf(e - m[gid]);
}

// edge pass 2: atomic add exp(e - m[dst])
__global__ void k_edge_sum(const int* __restrict__ ei, const float* __restrict__ als,
                           const float* __restrict__ ald, const float* __restrict__ m,
                           float* __restrict__ s) {
  int e = blockIdx.x * blockDim.x + threadIdx.x;
  if (e >= N_EDGES) return;
  int src = ei[e], dst = ei[N_EDGES + e];
#pragma unroll
  for (int hh = 0; hh < HEADS; hh++) {
    float v = als[src * 4 + hh] + ald[dst * 4 + hh];
    v = v > 0.f ? v : NEG * v;
    atomicAdd(&s[dst * 4 + hh], __expf(v - m[dst * 4 + hh]));
  }
}

// z init: self-loop contribution + gat_bias (no atomics)
__global__ void k_zinit(const float* __restrict__ xw, const float* __restrict__ als,
                        const float* __restrict__ ald, const float* __restrict__ m,
                        const float* __restrict__ s, const float* __restrict__ bias,
                        float* __restrict__ z) {
  int gid = blockIdx.x * blockDim.x + threadIdx.x;
  if (gid >= PMETA * N_NODES * 64) return;
  int p = gid / (N_NODES * 64);
  int r = gid % (N_NODES * 64);
  int n = r >> 6, q = r & 63;
  int c0 = q * 4, hh = q >> 4;
  int ih = (p * N_NODES + n) * 4 + hh;
  float e = als[ih] + ald[ih];
  e = e > 0.f ? e : NEG * e;
  float coef = __expf(e - m[ih]) / s[ih];
  size_t base = ((size_t)p * N_NODES + n) * HF + c0;
  float4 x = *(const float4*)(xw + base);
  const float* bs = bias + (size_t)p * HF + c0;
  float4 o;
  o.x = coef * x.x + bs[0];
  o.y = coef * x.y + bs[1];
  o.z = coef * x.z + bs[2];
  o.w = coef * x.w + bs[3];
  *(float4*)(z + base) = o;
}

// edge pass 3: weighted scatter, 64 threads/edge (pointers pre-offset per meta-path)
__global__ void __launch_bounds__(256)
k_edge_scatter(const int* __restrict__ ei, const float* __restrict__ als,
               const float* __restrict__ ald, const float* __restrict__ m,
               const float* __restrict__ s, const float* __restrict__ xw,
               float* __restrict__ z) {
  int t = threadIdx.x;
  int e = blockIdx.x * 4 + (t >> 6);
  if (e >= N_EDGES) return;
  int lane64 = t & 63;
  int src = ei[e], dst = ei[N_EDGES + e];
  int hh = lane64 >> 4;
  int ih = dst * 4 + hh;
  float v = als[src * 4 + hh] + ald[ih];
  v = v > 0.f ? v : NEG * v;
  float coef = __expf(v - m[ih]) / s[ih];
  int c0 = lane64 * 4;
  float4 x = *(const float4*)(xw + (size_t)src * HF + c0);
  float* zp = z + (size_t)dst * HF + c0;
  atomicAdd(zp + 0, coef * x.x);
  atomicAdd(zp + 1, coef * x.y);
  atomicAdd(zp + 2, coef * x.z);
  atomicAdd(zp + 3, coef * x.w);
}

// semantic GEMM: t = z_flat @ sa_w1, fused tanh(.+b1).w2 epilogue + mean accumulation
__global__ void __launch_bounds__(256)
k_gemm_sem(const float* __restrict__ zf, const uint32_t* __restrict__ w1pk,
           const float* __restrict__ b1, const float* __restrict__ w2,
           float* __restrict__ wsum) {
  __shared__ float rowsum[128];
  const int wid = threadIdx.x >> 5;
  const int lane = threadIdx.x & 31;
  if (threadIdx.x < 128) rowsum[threadIdx.x] = 0.f;
  __syncthreads();
  const int m0 = blockIdx.x * 128 + wid * 16;

  int arow_i = m0 + (lane & 15);
  arow_i = arow_i < MTOT ? arow_i : MTOT - 1;         // clamp: OOB rows masked in epilogue
  const float* arow = zf + (size_t)arow_i * HF + ((lane < 16) ? 0 : 8);
  const int klo2 = (lane < 16) ? 0 : 4;

  v8f acc[8] = {};
  for (int k0 = 0; k0 < HF; k0 += 64) {
    v16h a0 = load_a_f32(arow, k0);
    gemm_step(a0, w1pk, 0, (k0 >> 1) + klo2, acc);
    v16h a1 = load_a_f32(arow, k0 + 32);
    gemm_step(a1, w1pk, 0, ((k0 + 32) >> 1) + klo2, acc);
  }

  const int rbase = m0 + ((lane < 16) ? 0 : 8);
  float rs[8];
#pragma unroll
  for (int i = 0; i < 8; i++) rs[i] = 0.f;
#pragma unroll
  for (int nt = 0; nt < 8; nt++) {
    int col = nt * 16 + (lane & 15);
    float bb = b1[col], ww = w2[col];
#pragma unroll
    for (int i = 0; i < 8; i++) {
      int row = rbase + i;
      rs[i] += (row < MTOT) ? tanhf(acc[nt][i] + bb) * ww : 0.f;
    }
  }
#pragma unroll
  for (int i = 0; i < 8; i++) {
    float v = rs[i];
    v += __shfl_xor(v, 1);
    v += __shfl_xor(v, 2);
    v += __shfl_xor(v, 4);
    v += __shfl_xor(v, 8);
    if ((lane & 15) == 0)
      atomicAdd(&rowsum[wid * 16 + ((lane < 16) ? 0 : 8) + i], v);
  }
  __syncthreads();
  if (threadIdx.x < 128) {
    int row = blockIdx.x * 128 + threadIdx.x;
    if (row < MTOT) atomicAdd(&wsum[row / N_NODES], rowsum[threadIdx.x]);
  }
}

// beta = softmax(mean_n w[n,p])
__global__ void k_beta(const float* __restrict__ wsum, float* __restrict__ beta) {
  if (threadIdx.x == 0 && blockIdx.x == 0) {
    float w0 = wsum[0] / (float)N_NODES;
    float w1 = wsum[1] / (float)N_NODES;
    float w2 = wsum[2] / (float)N_NODES;
    float mx = fmaxf(w0, fmaxf(w1, w2));
    float e0 = __expf(w0 - mx), e1 = __expf(w1 - mx), e2 = __expf(w2 - mx);
    float inv = 1.f / (e0 + e1 + e2);
    beta[0] = e0 * inv; beta[1] = e1 * inv; beta[2] = e2 * inv;
  }
}

// out[n] = sum_p beta[p] * z[p][n]
__global__ void k_combine(const float* __restrict__ z, const float* __restrict__ beta,
                          float* __restrict__ out) {
  int gid = blockIdx.x * blockDim.x + threadIdx.x;
  if (gid >= N_NODES * 64) return;
  size_t base = (size_t)gid * 4;
  float b0 = beta[0], b1 = beta[1], b2 = beta[2];
  const size_t S = (size_t)N_NODES * HF;
  float4 z0 = *(const float4*)(z + base);
  float4 z1 = *(const float4*)(z + S + base);
  float4 z2 = *(const float4*)(z + 2 * S + base);
  float4 o;
  o.x = b0 * z0.x + b1 * z1.x + b2 * z2.x;
  o.y = b0 * z0.y + b1 * z1.y + b2 * z2.y;
  o.z = b0 * z0.z + b1 * z1.z + b2 * z2.z;
  o.w = b0 * z0.w + b1 * z1.w + b2 * z2.w;
  *(float4*)(out + base) = o;
}

// ---------------- host launch ----------------
extern "C" void kernel_launch(void* const* d_in, const int* in_sizes, int n_in,
                              void* d_out, int out_size, void* d_ws, size_t ws_size,
                              hipStream_t stream) {
  const float* h_in   = (const float*)d_in[0];
  const int*   ei[3]  = { (const int*)d_in[1], (const int*)d_in[2], (const int*)d_in[3] };
  const float* gatW   = (const float*)d_in[4];
  const float* asrc   = (const float*)d_in[5];
  const float* adst   = (const float*)d_in[6];
  const float* gbias  = (const float*)d_in[7];
  const float* saw1   = (const float*)d_in[8];
  const float* sab1   = (const float*)d_in[9];
  const float* saw2   = (const float*)d_in[10];
  float* out = (float*)d_out;

  char* ws = (char*)d_ws;
  float*    xw   = (float*)(ws + XW_OFF);
  float*    z    = (float*)(ws + Z_OFF);
  float*    als  = (float*)(ws + AS_OFF);
  float*    ald  = (float*)(ws + AD_OFF);
  float*    mbuf = (float*)(ws + M_OFF);
  float*    sbuf = (float*)(ws + S_OFF);
  uint32_t* wpk  = (uint32_t*)(ws + WPK_OFF);
  uint32_t* w1pk = (uint32_t*)(ws + W1PK_OFF);
  float*    wsum = (float*)(ws + WSUM_OFF);
  float*    beta = (float*)(ws + BETA_OFF);

  // 1) pack weights to f16 B-fragment layout + zero wsum
  {
    int tot = PMETA * HF * 128 + HID * 128 + 3;
    k_pack<<<(tot + 255) / 256, 256, 0, stream>>>(gatW, saw1, wpk, w1pk, wsum);
  }
  // 2) xw = h @ W[p]  (WMMA)
  k_gemm_xw<<<dim3((N_NODES + 127) / 128, PMETA * 2), 256, 0, stream>>>(h_in, wpk, xw);
  // 3) attention logit halves
  {
    int tot = PMETA * N_NODES * HEADS;
    k_alpha<<<(tot + 255) / 256, 256, 0, stream>>>(xw, asrc, adst, als, ald);
    k_minit<<<(tot + 255) / 256, 256, 0, stream>>>(als, ald, mbuf);
  }
  // 4) segment softmax, 3 passes (self-loops folded into node kernels)
  const int eg = (N_EDGES + 255) / 256;
  const size_t pnh = (size_t)N_NODES * HEADS;
  const size_t pnc = (size_t)N_NODES * HF;
  for (int p = 0; p < PMETA; p++)
    k_edge_max<<<eg, 256, 0, stream>>>(ei[p], als + p * pnh, ald + p * pnh, mbuf + p * pnh);
  {
    int tot = PMETA * N_NODES * HEADS;
    k_sinit<<<(tot + 255) / 256, 256, 0, stream>>>(als, ald, mbuf, sbuf);
  }
  for (int p = 0; p < PMETA; p++)
    k_edge_sum<<<eg, 256, 0, stream>>>(ei[p], als + p * pnh, ald + p * pnh,
                                       mbuf + p * pnh, sbuf + p * pnh);
  {
    int tot = PMETA * N_NODES * 64;
    k_zinit<<<(tot + 255) / 256, 256, 0, stream>>>(xw, als, ald, mbuf, sbuf, gbias, z);
  }
  for (int p = 0; p < PMETA; p++)
    k_edge_scatter<<<(N_EDGES + 3) / 4, 256, 0, stream>>>(
        ei[p], als + p * pnh, ald + p * pnh, mbuf + p * pnh, sbuf + p * pnh,
        xw + p * pnc, z + p * pnc);
  // 5) semantic attention (WMMA GEMM + fused tanh/dot/mean)
  k_gemm_sem<<<(MTOT + 127) / 128, 256, 0, stream>>>(z, w1pk, sab1, saw2, wsum);
  k_beta<<<1, 32, 0, stream>>>(wsum, beta);
  // 6) weighted combine over meta-paths
  k_combine<<<(N_NODES * 64 + 255) / 256, 256, 0, stream>>>(z, beta, out);
}
